// AttentionBlock_30820685316321
// MI455X (gfx1250) — matrix operand order repaired and checked by reading, not compiled, not run on hardware
//
#include <hip/hip_runtime.h>
#include <hip/hip_bf16.h>
#include <stdint.h>

typedef __attribute__((ext_vector_type(16))) __bf16 v16bf;
typedef __attribute__((ext_vector_type(8)))  __bf16 v8bf;
typedef __attribute__((ext_vector_type(4)))  __bf16 v4bf;
typedef __attribute__((ext_vector_type(8)))  float  v8f;

union FragBf { v16bf v; v8bf h[2]; };

__device__ __forceinline__ v8f wmma_bf16(const FragBf& a, const FragBf& b, v8f c) {
  return __builtin_amdgcn_wmma_f32_16x16x32_bf16(false, a.v, false, b.v, (short)0, c, false, false);
}

__device__ __forceinline__ float redmax16(float x) {
  #pragma unroll
  for (int m = 8; m >= 1; m >>= 1) x = fmaxf(x, __shfl_xor(x, m, 16));
  return x;
}
__device__ __forceinline__ float redsum16(float x) {
  #pragma unroll
  for (int m = 8; m >= 1; m >>= 1) x += __shfl_xor(x, m, 16);
  return x;
}

#define LDA 40    // 32 + 8 pad (bf16 elems), 80 B rows (16B aligned)
#define LDB 136   // 128 + 8 pad, 272 B rows
#define QK_SCALE 0.35355339059327373f  // 64^-0.25

// ---------------------------------------------------------------------------
// Kernel 1: qkv = x @ Wqkv + b ; scatter to Q (scaled, [bh][t][hc]),
// Kt (scaled, transposed [bh][hc][t]), V ([bh][t][hc]) as bf16.
// M=8192, K=512, N=1536. Block tile 128x128, 8 waves of 2x4 WMMA tiles.
// ---------------------------------------------------------------------------
__global__ __launch_bounds__(256)
void qkv_proj_kernel(const float* __restrict__ x, const float* __restrict__ w,
                     const float* __restrict__ bias,
                     __bf16* __restrict__ qws, __bf16* __restrict__ ktws,
                     __bf16* __restrict__ vws)
{
  __shared__ __align__(16) __bf16 aLds[128 * LDA];
  __shared__ __align__(16) __bf16 bLds[32 * LDB];

  const int tid  = threadIdx.x;
  const int lane = tid & 31;
  const int wv   = tid >> 5;
  const int wm   = wv & 3, wn = wv >> 2;     // 4x2 wave grid
  const int lo   = lane & 15, hi = lane >> 4;
  const int mBase = blockIdx.x * 128;
  const int nBase = blockIdx.y * 128;

  const v8f vzero = {0.f,0.f,0.f,0.f,0.f,0.f,0.f,0.f};
  v8f acc[2][4];
  #pragma unroll
  for (int i = 0; i < 2; ++i)
    #pragma unroll
    for (int j = 0; j < 4; ++j) acc[i][j] = vzero;

  for (int k0 = 0; k0 < 512; k0 += 32) {
    __syncthreads();
    #pragma unroll
    for (int i = 0; i < 4; ++i) {             // A tile: 128x32 f32 -> bf16
      int c = tid + i * 256;
      int row = c >> 3, col4 = c & 7;
      const float4 f = *(const float4*)(x + (size_t)(mBase + row) * 512 + k0 + col4 * 4);
      v4bf b4 = { (__bf16)f.x, (__bf16)f.y, (__bf16)f.z, (__bf16)f.w };
      *(v4bf*)(aLds + row * LDA + col4 * 4) = b4;
    }
    #pragma unroll
    for (int i = 0; i < 4; ++i) {             // B tile: 32x128 f32 -> bf16
      int c = tid + i * 256;
      int row = c >> 5, col4 = c & 31;
      const float4 f = *(const float4*)(w + (size_t)(k0 + row) * 1536 + nBase + col4 * 4);
      v4bf b4 = { (__bf16)f.x, (__bf16)f.y, (__bf16)f.z, (__bf16)f.w };
      *(v4bf*)(bLds + row * LDB + col4 * 4) = b4;
    }
    __syncthreads();

    FragBf aF[2], bF[4];
    #pragma unroll
    for (int tm = 0; tm < 2; ++tm) {
      const __bf16* p = aLds + (wm * 32 + tm * 16 + lo) * LDA;
      aF[tm].h[0] = *(const v8bf*)(p + hi * 8);
      aF[tm].h[1] = *(const v8bf*)(p + 16 + hi * 8);
    }
    #pragma unroll
    for (int tn = 0; tn < 4; ++tn) {
      const __bf16* p = bLds + wn * 64 + tn * 16 + lo;
      #pragma unroll
      for (int s = 0; s < 8; ++s) {
        bF[tn].v[s]     = p[(hi * 8 + s) * LDB];
        bF[tn].v[8 + s] = p[(16 + hi * 8 + s) * LDB];
      }
    }
    #pragma unroll
    for (int tm = 0; tm < 2; ++tm)
      #pragma unroll
      for (int tn = 0; tn < 4; ++tn)
        acc[tm][tn] = wmma_bf16(aF[tm], bF[tn], acc[tm][tn]);
  }

  // Epilogue: bias add + scatter to Q / Kt / V (bf16)
  #pragma unroll
  for (int tm = 0; tm < 2; ++tm)
    #pragma unroll
    for (int tn = 0; tn < 4; ++tn) {
      const int gn = nBase + wn * 64 + tn * 16 + lo;
      const float bia = bias[gn];
      const int which = gn >> 9;        // 0=q,1=k,2=v
      const int cc = gn & 511;
      const int h = cc >> 6, ch = cc & 63;
      #pragma unroll
      for (int v = 0; v < 8; ++v) {
        const int gm = mBase + wm * 32 + tm * 16 + v + 8 * hi;
        const int b = gm >> 10, t = gm & 1023;
        const int bh = b * 8 + h;
        const float val = acc[tm][tn][v] + bia;
        if (which == 0)
          qws[((size_t)bh * 1024 + t) * 64 + ch] = (__bf16)(val * QK_SCALE);
        else if (which == 1)
          ktws[((size_t)bh * 64 + ch) * 1024 + t] = (__bf16)(val * QK_SCALE);
        else
          vws[((size_t)bh * 1024 + t) * 64 + ch] = (__bf16)val;
      }
    }
}

// ---------------------------------------------------------------------------
// Kernel 2: flash attention per (b,h). Block = 8 waves x 16 queries = 128 q.
// Keys streamed in 128-key LDS chunks (Kt [64][128], V [128][64]).
// S tiles via WMMA, online softmax in C-layout, P -> LDS -> A-frag -> P@V.
// ---------------------------------------------------------------------------
#define LDKT 136  // 128 + 8
#define LDV  72   // 64 + 8

__global__ __launch_bounds__(256)
void flash_attn_kernel(const __bf16* __restrict__ qws, const __bf16* __restrict__ ktws,
                       const __bf16* __restrict__ vws, __bf16* __restrict__ aws)
{
  __shared__ __align__(16) __bf16 ktLds[64 * LDKT];
  __shared__ __align__(16) __bf16 vLds[128 * LDV];
  __shared__ __align__(16) __bf16 pLds[8 * 16 * 32];

  const int tid  = threadIdx.x;
  const int lane = tid & 31;
  const int wv   = tid >> 5;
  const int lo   = lane & 15, hi = lane >> 4;
  const int bh   = blockIdx.x >> 3;
  const int qc   = blockIdx.x & 7;
  const int t0   = qc * 128 + wv * 16;

  const __bf16* Q  = qws  + (size_t)bh * 1024 * 64;
  const __bf16* KT = ktws + (size_t)bh * 64 * 1024;
  const __bf16* V  = vws  + (size_t)bh * 1024 * 64;

  // Q A-fragments (16 queries x 64 hc = two k-steps), held in registers
  FragBf qF[2];
  #pragma unroll
  for (int kc = 0; kc < 2; ++kc) {
    const __bf16* p = Q + (size_t)(t0 + lo) * 64 + kc * 32 + hi * 8;
    qF[kc].h[0] = *(const v8bf*)(p);
    qF[kc].h[1] = *(const v8bf*)(p + 16);
  }

  const v8f vzero = {0.f,0.f,0.f,0.f,0.f,0.f,0.f,0.f};
  v8f o[4]; o[0] = o[1] = o[2] = o[3] = vzero;
  float mrow[8], lrow[8];
  #pragma unroll
  for (int v = 0; v < 8; ++v) { mrow[v] = -1e30f; lrow[v] = 0.f; }

  __bf16* pScr = pLds + wv * (16 * 32);

  for (int c0 = 0; c0 < 1024; c0 += 128) {
    __syncthreads();
    #pragma unroll
    for (int i = 0; i < 4; ++i) {            // Kt chunk: 64 x 128 bf16
      int c = tid + i * 256;
      int row = c >> 4, col8 = c & 15;
      *(uint4*)(ktLds + row * LDKT + col8 * 8) =
          *(const uint4*)(KT + (size_t)row * 1024 + c0 + col8 * 8);
    }
    #pragma unroll
    for (int i = 0; i < 4; ++i) {            // V chunk: 128 x 64 bf16
      int c = tid + i * 256;
      int row = c >> 3, col8 = c & 7;
      *(uint4*)(vLds + row * LDV + col8 * 8) =
          *(const uint4*)(V + (size_t)(c0 + row) * 64 + col8 * 8);
    }
    __syncthreads();

    for (int kb = 0; kb < 4; ++kb) {         // 32 keys per step
      v8f s0 = vzero, s1 = vzero;
      #pragma unroll
      for (int kc = 0; kc < 2; ++kc) {       // contraction over hc=64
        FragBf kB;
        const __bf16* p = ktLds + (kc * 32) * LDKT + kb * 32 + lo;
        #pragma unroll
        for (int s = 0; s < 8; ++s) { kB.v[s] = p[(hi * 8 + s) * LDKT]; kB.v[8 + s] = p[(16 + hi * 8 + s) * LDKT]; }
        s0 = wmma_bf16(qF[kc], kB, s0);
        #pragma unroll
        for (int s = 0; s < 8; ++s) { kB.v[s] = p[(hi * 8 + s) * LDKT + 16]; kB.v[8 + s] = p[(16 + hi * 8 + s) * LDKT + 16]; }
        s1 = wmma_bf16(qF[kc], kB, s1);
      }

      // online softmax over this 32-key block (rows live in wave halves)
      #pragma unroll
      for (int v = 0; v < 8; ++v) {
        float mx = redmax16(fmaxf(s0[v], s1[v]));
        float mn = fmaxf(mrow[v], mx);
        float al = __expf(mrow[v] - mn);
        float e0 = __expf(s0[v] - mn);
        float e1 = __expf(s1[v] - mn);
        lrow[v] = lrow[v] * al + redsum16(e0 + e1);
        mrow[v] = mn;
        o[0][v] *= al; o[1][v] *= al; o[2][v] *= al; o[3][v] *= al;
        pScr[(v + 8 * hi) * 32 + lo]      = (__bf16)e0;   // C-layout -> [16][32]
        pScr[(v + 8 * hi) * 32 + lo + 16] = (__bf16)e1;
      }
      // per-wave LDS is in-order; wait before re-reading as an A-fragment
      asm volatile("s_wait_dscnt 0" ::: "memory");

      FragBf aP;
      {
        const __bf16* p = pScr + lo * 32 + hi * 8;
        aP.h[0] = *(const v8bf*)(p);
        aP.h[1] = *(const v8bf*)(p + 16);
      }
      #pragma unroll
      for (int tn = 0; tn < 4; ++tn) {       // O += P @ V
        FragBf vB;
        const __bf16* p = vLds + (kb * 32) * LDV + tn * 16 + lo;
        #pragma unroll
        for (int s = 0; s < 8; ++s) {
          vB.v[s]     = p[(hi * 8 + s) * LDV];
          vB.v[8 + s] = p[(16 + hi * 8 + s) * LDV];
        }
        o[tn] = wmma_bf16(aP, vB, o[tn]);
      }
    }
  }

  // Normalize + store A in [B*T][C] bf16 (row = b*1024+t, col = h*64+hc)
  const int b = bh >> 3, h = bh & 7;
  #pragma unroll
  for (int v = 0; v < 8; ++v) {
    const float inv = 1.0f / lrow[v];
    const int trow = t0 + v + 8 * hi;
    const size_t base = (size_t)(b * 1024 + trow) * 512 + h * 64 + lo;
    #pragma unroll
    for (int tn = 0; tn < 4; ++tn)
      aws[base + tn * 16] = (__bf16)(o[tn][v] * inv);
  }
}

// ---------------------------------------------------------------------------
// Kernel 3: out = A @ Wout + b_out. M=8192, K=512, N=512; fp32 output.
// ---------------------------------------------------------------------------
__global__ __launch_bounds__(256)
void out_proj_kernel(const __bf16* __restrict__ aws, const float* __restrict__ w,
                     const float* __restrict__ bias, float* __restrict__ out)
{
  __shared__ __align__(16) __bf16 aLds[128 * LDA];
  __shared__ __align__(16) __bf16 bLds[32 * LDB];

  const int tid  = threadIdx.x;
  const int lane = tid & 31;
  const int wv   = tid >> 5;
  const int wm   = wv & 3, wn = wv >> 2;
  const int lo   = lane & 15, hi = lane >> 4;
  const int mBase = blockIdx.x * 128;
  const int nBase = blockIdx.y * 128;

  const v8f vzero = {0.f,0.f,0.f,0.f,0.f,0.f,0.f,0.f};
  v8f acc[2][4];
  #pragma unroll
  for (int i = 0; i < 2; ++i)
    #pragma unroll
    for (int j = 0; j < 4; ++j) acc[i][j] = vzero;

  for (int k0 = 0; k0 < 512; k0 += 32) {
    __syncthreads();
    #pragma unroll
    for (int i = 0; i < 2; ++i) {            // A tile: 128x32 bf16 copy
      int c = tid + i * 256;
      int row = c >> 2, col8 = c & 3;
      *(uint4*)(aLds + row * LDA + col8 * 8) =
          *(const uint4*)(aws + (size_t)(mBase + row) * 512 + k0 + col8 * 8);
    }
    #pragma unroll
    for (int i = 0; i < 4; ++i) {            // B tile: 32x128 f32 -> bf16
      int c = tid + i * 256;
      int row = c >> 5, col4 = c & 31;
      const float4 f = *(const float4*)(w + (size_t)(k0 + row) * 512 + nBase + col4 * 4);
      v4bf b4 = { (__bf16)f.x, (__bf16)f.y, (__bf16)f.z, (__bf16)f.w };
      *(v4bf*)(bLds + row * LDB + col4 * 4) = b4;
    }
    __syncthreads();

    FragBf aF[2], bF[4];
    #pragma unroll
    for (int tm = 0; tm < 2; ++tm) {
      const __bf16* p = aLds + (wm * 32 + tm * 16 + lo) * LDA;
      aF[tm].h[0] = *(const v8bf*)(p + hi * 8);
      aF[tm].h[1] = *(const v8bf*)(p + 16 + hi * 8);
    }
    #pragma unroll
    for (int tn = 0; tn < 4; ++tn) {
      const __bf16* p = bLds + wn * 64 + tn * 16 + lo;
      #pragma unroll
      for (int s = 0; s < 8; ++s) {
        bF[tn].v[s]     = p[(hi * 8 + s) * LDB];
        bF[tn].v[8 + s] = p[(16 + hi * 8 + s) * LDB];
      }
    }
    #pragma unroll
    for (int tm = 0; tm < 2; ++tm)
      #pragma unroll
      for (int tn = 0; tn < 4; ++tn)
        acc[tm][tn] = wmma_bf16(aF[tm], bF[tn], acc[tm][tn]);
  }

  #pragma unroll
  for (int tm = 0; tm < 2; ++tm)
    #pragma unroll
    for (int tn = 0; tn < 4; ++tn) {
      const int gn = nBase + wn * 64 + tn * 16 + lo;
      const float bia = bias[gn];
      #pragma unroll
      for (int v = 0; v < 8; ++v) {
        const int gm = mBase + wm * 32 + tm * 16 + v + 8 * hi;
        out[(size_t)gm * 512 + gn] = acc[tm][tn][v] + bia;
      }
    }
}

// ---------------------------------------------------------------------------
extern "C" void kernel_launch(void* const* d_in, const int* in_sizes, int n_in,
                              void* d_out, int out_size, void* d_ws, size_t ws_size,
                              hipStream_t stream) {
  const float* x      = (const float*)d_in[0];   // [8,32,32,512]
  const float* qkv_w  = (const float*)d_in[1];   // [512,1536]
  const float* qkv_b  = (const float*)d_in[2];   // [1536]
  const float* out_w  = (const float*)d_in[3];   // [512,512]
  const float* out_b  = (const float*)d_in[4];   // [512]
  float* out          = (float*)d_out;           // [8192,512]

  const size_t per = (size_t)8 * 1024 * 512;     // 4,194,304 bf16 elems each
  __bf16* qws  = (__bf16*)d_ws;
  __bf16* ktws = qws + per;
  __bf16* vws  = ktws + per;
  __bf16* aws  = vws + per;

  dim3 g1(64, 12);
  qkv_proj_kernel<<<g1, 256, 0, stream>>>(x, qkv_w, qkv_b, qws, ktws, vws);

  flash_attn_kernel<<<512, 256, 0, stream>>>(qws, ktws, vws, aws);

  dim3 g3(64, 4);
  out_proj_kernel<<<g3, 256, 0, stream>>>(aws, out_w, out_b, out);
}
